// _EoMT_24635932410446
// MI455X (gfx1250) — compile-verified, matrix operand order checked by reading, not run on hardware
//
#include <hip/hip_runtime.h>
#include <hip/hip_bf16.h>
#include <math.h>

// ---------------- problem constants ----------------
#define BB 2
#define CC 768
#define LL 12
#define NHEADS 12
#define DHD 64
#define NQ 100
#define PREF 5
#define NPATCH 1024
#define N1 (PREF + NPATCH)        // 1029
#define N2 (NQ + PREF + NPATCH)   // 1129
#define H3 3072
#define NCLS1 151
#define QCOL (NQ + PREF)          // 105
#define NPMAX 1152                // N2 rounded up to 32

typedef __bf16 bf16_t;
typedef __bf16 v16bf __attribute__((ext_vector_type(16)));
typedef __bf16 v8bf  __attribute__((ext_vector_type(8)));
typedef float  v8f   __attribute__((ext_vector_type(8)));

static __device__ __forceinline__ float gelu_f(float v) {
  return 0.5f * v * (1.0f + erff(v * 0.7071067811865475f));
}

// ---- CDNA5 wave32 bf16 A/B fragment load (16x32 tile of row-major [rows x ld]).
// Layout per ISA 7.12.2: lanes 0-15 hold row m, halves j<8 -> K=kb+j, j>=8 -> K=kb+16+(j-8);
// lanes 16-31 same rows with K offset +8.
// Branch-free: rows are CLAMPED (out-of-range rows feed only discarded outputs);
// caller guarantees K is a multiple of 32 and rows are 16B-aligned (ld % 8 == 0).
static __device__ __forceinline__ v16bf load_frag(const bf16_t* __restrict__ base,
                                                  int row0, int ld, int kb,
                                                  int rowClamp, int lane) {
  int r  = row0 + (lane & 15);
  r = (r < rowClamp) ? r : rowClamp;            // rowClamp = rows-1
  int k0 = kb + ((lane & 16) ? 8 : 0);
  const bf16_t* rp = base + (size_t)r * (size_t)ld + k0;
  v8bf lo = *(const v8bf*)rp;                   // K = k0 .. k0+7
  v8bf hi = *(const v8bf*)(rp + 16);            // K = k0+16 .. k0+23
  v16bf f;
#pragma unroll
  for (int j = 0; j < 8; ++j) { f[j] = lo[j]; f[j + 8] = hi[j]; }
  return f;
}

// ---- generic batched WMMA GEMM: Out[m,n] = sum_k A[m,k]*W[n,k] (+bias[n]) (+gelu)
// grid: x = N/64 tiles, y = M/128 (4 waves x 32 rows), z = batch. K % 32 == 0.
__global__ void __launch_bounds__(128)
k_gemm(const bf16_t* __restrict__ A, const bf16_t* __restrict__ Wm,
       const float* __restrict__ bias, float* __restrict__ outF,
       bf16_t* __restrict__ outH, const int* __restrict__ rowMap,
       int M, int N, int K, int ldA, int ldW, int ldO,
       long long sA, long long sW, long long sO, int act)
{
  int z = blockIdx.z;
  const bf16_t* Az = A + (size_t)z * (size_t)sA;
  const bf16_t* Wz = Wm + (size_t)z * (size_t)sW;
  size_t oz = (size_t)z * (size_t)sO;
  int wave = threadIdx.x >> 5;
  int lane = threadIdx.x & 31;
  int m0 = blockIdx.y * 128 + wave * 32;
  int n0 = blockIdx.x * 64;
  if (m0 >= M) return;
  int mc = M - 1, nc = N - 1;

  v8f acc[2][4];
#pragma unroll
  for (int u = 0; u < 2; ++u)
#pragma unroll
    for (int t = 0; t < 4; ++t)
#pragma unroll
      for (int r = 0; r < 8; ++r) acc[u][t][r] = 0.0f;

  for (int kb = 0; kb < K; kb += 32) {
    v16bf a0 = load_frag(Az, m0,      ldA, kb, mc, lane);
    v16bf a1 = load_frag(Az, m0 + 16, ldA, kb, mc, lane);
#pragma unroll
    for (int t = 0; t < 4; ++t) {
      v16bf b = load_frag(Wz, n0 + 16 * t, ldW, kb, nc, lane);
      acc[0][t] = __builtin_amdgcn_wmma_f32_16x16x32_bf16(false, a0, false, b,
                                                          (short)0, acc[0][t], false, false);
      acc[1][t] = __builtin_amdgcn_wmma_f32_16x16x32_bf16(false, a1, false, b,
                                                          (short)0, acc[1][t], false, false);
    }
  }
  // C/D layout: lane -> N (lane&15), VGPR r -> M = r (+8 for lanes 16-31)
  int mlo = (lane & 16) ? 8 : 0;
  int nl  = lane & 15;
#pragma unroll
  for (int t = 0; t < 4; ++t) {
    int n = n0 + 16 * t + nl;
    if (n >= N) continue;
    float bv = bias ? bias[n] : 0.0f;
#pragma unroll
    for (int u = 0; u < 2; ++u) {
#pragma unroll
      for (int r = 0; r < 8; ++r) {
        int m = m0 + 16 * u + mlo + r;
        if (m >= M) continue;
        float v = acc[u][t][r] + bv;
        if (act == 1) v = gelu_f(v);
        size_t row = rowMap ? (size_t)rowMap[m] : (size_t)m;
        size_t o = oz + row * (size_t)ldO + (size_t)n;
        if (outF) outF[o] = v;
        if (outH) outH[o] = (bf16_t)v;
      }
    }
  }
}

// ---- layernorm over C=768 (block per row), optional bf16/f32 outputs
__global__ void __launch_bounds__(256)
k_layernorm(const float* __restrict__ X, const float* __restrict__ w,
            const float* __restrict__ bvec, bf16_t* __restrict__ outH,
            float* __restrict__ outF)
{
  int row = blockIdx.x;
  const float* xr = X + (size_t)row * CC;
  __shared__ float red[256];
  int t = threadIdx.x;
  float v[3]; float s = 0.0f;
#pragma unroll
  for (int j = 0; j < 3; ++j) { v[j] = xr[t + 256 * j]; s += v[j]; }
  red[t] = s; __syncthreads();
  for (int q = 128; q > 0; q >>= 1) { if (t < q) red[t] += red[t + q]; __syncthreads(); }
  float mean = red[0] * (1.0f / CC); __syncthreads();
  s = 0.0f;
#pragma unroll
  for (int j = 0; j < 3; ++j) { float d = v[j] - mean; s += d * d; }
  red[t] = s; __syncthreads();
  for (int q = 128; q > 0; q >>= 1) { if (t < q) red[t] += red[t + q]; __syncthreads(); }
  float rstd = rsqrtf(red[0] * (1.0f / CC) + 1e-6f);
#pragma unroll
  for (int j = 0; j < 3; ++j) {
    int c = t + 256 * j;
    float y = (v[j] - mean) * rstd * w[c] + bvec[c];
    if (outH) outH[(size_t)row * CC + c] = (bf16_t)y;
    if (outF) outF[(size_t)row * CC + c] = y;
  }
}

// ---- masked softmax row -> bf16 probabilities; zero-fills pad columns [N, Np)
__global__ void __launch_bounds__(256)
k_softmax(const float* __restrict__ S, bf16_t* __restrict__ Aout,
          int N, int Np, int useMask, const unsigned char* __restrict__ qmask)
{
  int idx = blockIdx.x;
  int z = idx / N;            // b*NHEADS + h
  int i = idx - z * N;
  int b = z / NHEADS;
  size_t base = ((size_t)z * Np + i) * (size_t)Np;
  __shared__ float red[256];
  int t = threadIdx.x;
  bool mrow = (useMask && i < NQ);
  const unsigned char* qm = mrow ? (qmask + ((size_t)b * NQ + i) * NPATCH) : nullptr;
  float mx = -3.0e38f;
  for (int j = t; j < N; j += 256) {
    float v = S[base + j];
    if (mrow && j >= QCOL && !qm[j - QCOL]) v = -3.0e38f;
    mx = fmaxf(mx, v);
  }
  red[t] = mx; __syncthreads();
  for (int q = 128; q > 0; q >>= 1) { if (t < q) red[t] = fmaxf(red[t], red[t + q]); __syncthreads(); }
  mx = red[0]; __syncthreads();
  float sum = 0.0f;
  for (int j = t; j < N; j += 256) {
    float v = S[base + j];
    bool ok = !(mrow && j >= QCOL && !qm[j - QCOL]);
    sum += ok ? __expf(v - mx) : 0.0f;
  }
  red[t] = sum; __syncthreads();
  for (int q = 128; q > 0; q >>= 1) { if (t < q) red[t] += red[t + q]; __syncthreads(); }
  float inv = 1.0f / red[0];
  for (int j = t; j < Np; j += 256) {
    float e = 0.0f;
    if (j < N) {
      float v = S[base + j];
      bool ok = !(mrow && j >= QCOL && !qm[j - QCOL]);
      e = ok ? __expf(v - mx) * inv : 0.0f;
    }
    Aout[base + j] = (bf16_t)e;
  }
}

// ---------------- elementwise kernels ----------------
__global__ void k_convert(const float* __restrict__ s, bf16_t* __restrict__ d, long long n) {
  long long i = (long long)blockIdx.x * 256 + threadIdx.x;
  if (i < n) d[i] = (bf16_t)s[i];
}

__global__ void k_pack_uptw(const float* __restrict__ uptw, bf16_t* __restrict__ dst, int total) {
  int idx = blockIdx.x * 256 + threadIdx.x;
  if (idx >= total) return;
  int c = idx % CC; int r = idx / CC;
  int o = r % CC; int r2 = r / CC;
  int dx = r2 % 4; int s = r2 / 4;
  // src: upt_w[s, c, o, d, x]
  dst[idx] = (bf16_t)uptw[(((size_t)s * CC + c) * CC + o) * 4 + dx];
}

__global__ void k_patchify(const float* __restrict__ x, const float* __restrict__ mean,
                           const float* __restrict__ stdv, bf16_t* __restrict__ P0, int total) {
  int idx = blockIdx.x * 256 + threadIdx.x;
  if (idx >= total) return;
  int e = idx % CC; int rest = idx / CC;
  int patch = rest % NPATCH; int b = rest / NPATCH;
  int c = e >> 8; int r2 = e & 255; int py = r2 >> 4; int px = r2 & 15;
  int gy = patch >> 5; int gx = patch & 31;
  float pix = x[(((size_t)b * 3 + c) * 512 + (gy * 16 + py)) * 512 + (gx * 16 + px)];
  P0[idx] = (bf16_t)((pix * (1.0f / 255.0f) - mean[c]) / stdv[c]);
}

__global__ void k_init_x(const float* __restrict__ pe, const float* __restrict__ pos,
                         const float* __restrict__ prefix, float* __restrict__ X, int total) {
  int idx = blockIdx.x * 256 + threadIdx.x;
  if (idx >= total) return;
  int c = idx % CC; int r = idx / CC; int n = r % N1; int b = r / N1;
  float v;
  if (n < PREF) v = prefix[n * CC + c];
  else { int p = n - PREF; v = pe[((size_t)b * NPATCH + p) * CC + c] + pos[(size_t)p * CC + c]; }
  X[idx] = v;
}

__global__ void k_insert_q(const float* __restrict__ Xold, const float* __restrict__ qemb,
                           float* __restrict__ Xnew, int total) {
  int idx = blockIdx.x * 256 + threadIdx.x;
  if (idx >= total) return;
  int c = idx % CC; int r = idx / CC; int n = r % N2; int b = r / N2;
  Xnew[idx] = (n < NQ) ? qemb[(size_t)n * CC + c]
                       : Xold[((size_t)b * N1 + (n - NQ)) * CC + c];
}

__global__ void k_qkv_split(const float* __restrict__ qkv, bf16_t* __restrict__ Q,
                            bf16_t* __restrict__ Kb, bf16_t* __restrict__ Vt,
                            int Ncur, int Np, int total) {
  int idx = blockIdx.x * 256 + threadIdx.x;
  if (idx >= total) return;
  int c = idx % CC; int r = idx / CC; int n = r % Ncur; int b = r / Ncur;
  int h = c >> 6; int d = c & 63;
  const float* row = qkv + (size_t)r * (3 * CC);
  size_t zo = (size_t)(b * NHEADS + h);
  Q [(zo * Ncur + n) * DHD + d] = (bf16_t)(row[c] * 0.125f);   // fold 1/sqrt(64)
  Kb[(zo * Ncur + n) * DHD + d] = (bf16_t)row[CC + c];
  Vt[(zo * DHD + d) * (size_t)Np + n] = (bf16_t)row[2 * CC + c];
}

// zero-fill Vt pad columns [Ncur, Np) so the O-GEMM can run with K = Np
__global__ void k_padvt(bf16_t* __restrict__ Vt, int Ncur, int Np, int total) {
  int idx = blockIdx.x * 256 + threadIdx.x;
  if (idx >= total) return;
  int pad = Np - Ncur;
  int col = idx % pad; int r = idx / pad;     // r = z*DHD + d
  Vt[(size_t)r * Np + Ncur + col] = (bf16_t)0.0f;
}

__global__ void k_merge(const float* __restrict__ O, bf16_t* __restrict__ ATT,
                        int Ncur, int total) {
  int idx = blockIdx.x * 256 + threadIdx.x;
  if (idx >= total) return;
  int c = idx % CC; int r = idx / CC; int n = r % Ncur; int b = r / Ncur;
  int h = c >> 6; int d = c & 63;
  ATT[idx] = (bf16_t)O[(((size_t)(b * NHEADS + h)) * Ncur + n) * DHD + d];
}

__global__ void k_residual(float* __restrict__ X, const float* __restrict__ Y,
                           const float* __restrict__ ls, int total) {
  int idx = blockIdx.x * 256 + threadIdx.x;
  if (idx >= total) return;
  X[idx] += ls[idx % CC] * Y[idx];
}

__global__ void k_copyF0(const bf16_t* __restrict__ XN, bf16_t* __restrict__ F0, int total) {
  int idx = blockIdx.x * 256 + threadIdx.x;
  if (idx >= total) return;
  int c = idx % CC; int r = idx / CC; int p = r % NPATCH; int b = r / NPATCH;
  F0[idx] = XN[((size_t)b * N2 + QCOL + p) * CC + c];
}

__global__ void k_fill_rowmap(int* __restrict__ maps, int Hh, int Ww, int total) {
  int idx = blockIdx.x * 256 + threadIdx.x;
  if (idx >= total) return;
  int M = total / 4;
  int m = idx % M; int dx = idx / M;
  int hw = Hh * Ww;
  int b = m / hw; int r = m % hw; int h = r / Ww; int w_ = r % Ww;
  int d = dx >> 1; int xx = dx & 1;
  maps[idx] = b * (4 * hw) + (2 * h + d) * (2 * Ww) + 2 * w_ + xx;
}

__global__ void k_dwconv(const float* __restrict__ in, const float* __restrict__ wgt,
                         float* __restrict__ out, int OH, int OW, int total) {
  int idx = blockIdx.x * 256 + threadIdx.x;
  if (idx >= total) return;
  int c = idx % CC; int r = idx / CC;
  int x_ = r % OW; r /= OW;
  int y_ = r % OH; int b = r / OH;
  float s = 0.0f;
#pragma unroll
  for (int ky = 0; ky < 3; ++ky) {
    int iy = y_ + ky - 1; if (iy < 0 || iy >= OH) continue;
#pragma unroll
    for (int kx = 0; kx < 3; ++kx) {
      int ix = x_ + kx - 1; if (ix < 0 || ix >= OW) continue;
      s += in[(((size_t)b * OH + iy) * OW + ix) * CC + c] * wgt[c * 9 + ky * 3 + kx];
    }
  }
  out[idx] = s;
}

// antialiased bilinear 128->32 downsample, threshold > 0 -> uint8 mask
__global__ void k_qmask(const float* __restrict__ mlog, unsigned char* __restrict__ qm, int total) {
  int idx = blockIdx.x * 256 + threadIdx.x;
  if (idx >= total) return;
  int gx = idx & 31; int r = idx >> 5;
  int gy = r & 31; r >>= 5;
  int q = r % NQ; int b = r / NQ;
  const float* mp = mlog + ((size_t)b * NQ + q) * 16384;
  float sy = 4.0f * gy + 1.5f, sx = 4.0f * gx + 1.5f;
  int yb = 4 * gy - 2, xb = 4 * gx - 2;
  float acc = 0.0f;
  for (int a = 0; a < 8; ++a) {
    int iy = yb + a; if (iy < 0 || iy > 127) continue;
    float wy = 1.0f - fabsf(iy - sy) * 0.25f; if (wy <= 0.0f) continue;
    for (int e = 0; e < 8; ++e) {
      int ix = xb + e; if (ix < 0 || ix > 127) continue;
      float wx = 1.0f - fabsf(ix - sx) * 0.25f; if (wx <= 0.0f) continue;
      acc += wy * wx * mp[iy * 128 + ix];
    }
  }
  qm[idx] = (acc > 0.0f) ? 1 : 0;
}

// ---------------- host orchestration ----------------
extern "C" void kernel_launch(void* const* d_in, const int* in_sizes, int n_in,
                              void* d_out, int out_size, void* d_ws, size_t ws_size,
                              hipStream_t stream) {
  (void)in_sizes; (void)n_in; (void)out_size; (void)ws_size;
  const float* x_in    = (const float*)d_in[0];
  const float* mean_in = (const float*)d_in[1];
  const float* std_in  = (const float*)d_in[2];
  const float* patchw  = (const float*)d_in[3];
  const float* patchb  = (const float*)d_in[4];
  const float* pos_in  = (const float*)d_in[5];
  const float* prefix  = (const float*)d_in[6];
  const float* qkvw    = (const float*)d_in[7];
  const float* qkvb    = (const float*)d_in[8];
  const float* projw   = (const float*)d_in[9];
  const float* projb   = (const float*)d_in[10];
  const float* ln1w    = (const float*)d_in[11];
  const float* ln1b    = (const float*)d_in[12];
  const float* ln2w    = (const float*)d_in[13];
  const float* ln2b    = (const float*)d_in[14];
  const float* ls1     = (const float*)d_in[15];
  const float* ls2     = (const float*)d_in[16];
  const float* fc1w    = (const float*)d_in[17];
  const float* fc1b    = (const float*)d_in[18];
  const float* fc2w    = (const float*)d_in[19];
  const float* fc2b    = (const float*)d_in[20];
  const float* normw   = (const float*)d_in[21];
  const float* normb   = (const float*)d_in[22];
  const float* qemb    = (const float*)d_in[23];
  const float* classw  = (const float*)d_in[24];
  const float* classb  = (const float*)d_in[25];
  const float* mh1w    = (const float*)d_in[26];
  const float* mh1b    = (const float*)d_in[27];
  const float* mh2w    = (const float*)d_in[28];
  const float* mh2b    = (const float*)d_in[29];
  const float* mh3w    = (const float*)d_in[30];
  const float* mh3b    = (const float*)d_in[31];
  const float* uptw    = (const float*)d_in[32];
  const float* uptb    = (const float*)d_in[33];
  const float* updw    = (const float*)d_in[34];
  const float* uplnw   = (const float*)d_in[35];
  const float* uplnb   = (const float*)d_in[36];
  float* outF = (float*)d_out;
  const size_t classBase = 5ull * BB * NQ * 16384;

  // ----- workspace arena -----
  char* wsb = (char*)d_ws;
  size_t off = 0;
  auto arena = [&](size_t bytes) -> void* {
    off = (off + 255) & ~(size_t)255;
    void* p = wsb + off; off += bytes; return p;
  };
  float*  X     = (float*)arena(4ull * BB * N2 * CC);
  float*  X2    = (float*)arena(4ull * BB * N2 * CC);
  bf16_t* XN    = (bf16_t*)arena(2ull * BB * N2 * CC);
  float*  TMP0  = (float*)arena(4ull * BB * N2 * 3 * CC);
  bf16_t* Qb    = (bf16_t*)arena(2ull * BB * NHEADS * N2 * DHD);
  bf16_t* Kbf   = (bf16_t*)arena(2ull * BB * NHEADS * N2 * DHD);
  bf16_t* Vt    = (bf16_t*)arena(2ull * BB * NHEADS * DHD * NPMAX);
  float*  Sbuf  = (float*)arena(4ull * BB * NHEADS * NPMAX * NPMAX);
  bf16_t* Ab    = (bf16_t*)arena(2ull * BB * NHEADS * NPMAX * NPMAX);
  float*  Obuf  = (float*)arena(4ull * BB * NHEADS * N2 * DHD);
  bf16_t* ATT   = (bf16_t*)arena(2ull * BB * N2 * CC);
  bf16_t* FC1H  = (bf16_t*)arena(2ull * BB * N2 * H3);
  bf16_t* P0    = (bf16_t*)arena(2ull * BB * NPATCH * CC);
  bf16_t* HM1   = (bf16_t*)arena(2ull * BB * NQ * CC);
  bf16_t* HM2   = (bf16_t*)arena(2ull * BB * NQ * CC);
  bf16_t* HM3   = (bf16_t*)arena(2ull * BB * NQ * CC);
  bf16_t* Fbuf  = (bf16_t*)arena(2ull * BB * (NPATCH + 4 * NPATCH + 16 * NPATCH) * CC);
  float*  CIN   = (float*)arena(4ull * BB * 128 * 128 * CC);
  float*  COUT  = (float*)arena(4ull * BB * 128 * 128 * CC);
  unsigned char* QM = (unsigned char*)arena((size_t)BB * NQ * NPATCH);
  int*    RMAP  = (int*)arena(4ull * 4 * BB * 64 * 64);
  bf16_t* qkvw_bf  = (bf16_t*)arena(2ull * LL * 3 * CC * CC);
  bf16_t* projw_bf = (bf16_t*)arena(2ull * LL * CC * CC);
  bf16_t* fc1w_bf  = (bf16_t*)arena(2ull * LL * H3 * CC);
  bf16_t* fc2w_bf  = (bf16_t*)arena(2ull * LL * CC * H3);
  bf16_t* patchw_bf= (bf16_t*)arena(2ull * CC * CC);
  bf16_t* classw_bf= (bf16_t*)arena(2ull * NCLS1 * CC);
  bf16_t* mh1_bf   = (bf16_t*)arena(2ull * CC * CC);
  bf16_t* mh2_bf   = (bf16_t*)arena(2ull * CC * CC);
  bf16_t* mh3_bf   = (bf16_t*)arena(2ull * CC * CC);
  bf16_t* uptw_bf  = (bf16_t*)arena(2ull * 2 * 4 * CC * CC);

  auto blocks = [](long long n) { return dim3((unsigned)((n + 255) / 256)); };
  auto conv = [&](const float* s, bf16_t* d, long long n) {
    k_convert<<<blocks(n), 256, 0, stream>>>(s, d, n);
  };
  auto gemm = [&](const bf16_t* A, const bf16_t* W, const float* bias,
                  float* oF, bf16_t* oH, const int* rmap,
                  int M, int N, int K, int ldA, int ldW, int ldO,
                  long long sA, long long sW, long long sO, int act, int nz) {
    dim3 g((unsigned)((N + 63) / 64), (unsigned)((M + 127) / 128), (unsigned)nz);
    k_gemm<<<g, 128, 0, stream>>>(A, W, bias, oF, oH, rmap,
                                  M, N, K, ldA, ldW, ldO, sA, sW, sO, act);
  };

  // ----- weight conversion (once per launch; deterministic) -----
  conv(qkvw,  qkvw_bf,  (long long)LL * 3 * CC * CC);
  conv(projw, projw_bf, (long long)LL * CC * CC);
  conv(fc1w,  fc1w_bf,  (long long)LL * H3 * CC);
  conv(fc2w,  fc2w_bf,  (long long)LL * CC * H3);
  conv(patchw, patchw_bf, (long long)CC * CC);
  conv(classw, classw_bf, (long long)NCLS1 * CC);
  conv(mh1w, mh1_bf, (long long)CC * CC);
  conv(mh2w, mh2_bf, (long long)CC * CC);
  conv(mh3w, mh3_bf, (long long)CC * CC);
  k_pack_uptw<<<blocks(2ll * 4 * CC * CC), 256, 0, stream>>>(uptw, uptw_bf, 2 * 4 * CC * CC);

  // ----- patch embed -----
  k_patchify<<<blocks((long long)BB * NPATCH * CC), 256, 0, stream>>>(x_in, mean_in, std_in, P0,
                                                                      BB * NPATCH * CC);
  gemm(P0, patchw_bf, patchb, TMP0, nullptr, nullptr,
       BB * NPATCH, CC, CC, CC, CC, CC, 0, 0, 0, 0, 1);
  k_init_x<<<blocks((long long)BB * N1 * CC), 256, 0, stream>>>(TMP0, pos_in, prefix, X,
                                                                BB * N1 * CC);

  // ----- predict head (stage 0..4), sequence length is N2 whenever called -----
  auto predict = [&](int stage) {
    k_layernorm<<<BB * N2, 256, 0, stream>>>(X2, normw, normb, XN, nullptr);
    // class logits -> d_out
    gemm(XN, classw_bf, classb,
         outF + classBase + (size_t)stage * BB * NQ * NCLS1, nullptr, nullptr,
         NQ, NCLS1, CC, CC, CC, NCLS1,
         (long long)N2 * CC, 0, (long long)NQ * NCLS1, 0, BB);
    // mask-head MLP
    gemm(XN, mh1_bf, mh1b, nullptr, HM1, nullptr,
         NQ, CC, CC, CC, CC, CC, (long long)N2 * CC, 0, (long long)NQ * CC, 1, BB);
    gemm(HM1, mh2_bf, mh2b, nullptr, HM2, nullptr,
         BB * NQ, CC, CC, CC, CC, CC, 0, 0, 0, 1, 1);
    gemm(HM2, mh3_bf, mh3b, nullptr, HM3, nullptr,
         BB * NQ, CC, CC, CC, CC, CC, 0, 0, 0, 0, 1);
    // F0 = patch tokens of LN(x), channel-last
    k_copyF0<<<blocks((long long)BB * NPATCH * CC), 256, 0, stream>>>(XN, Fbuf, BB * NPATCH * CC);
    bf16_t* Fs = Fbuf;
    bf16_t* Fn = Fbuf + (size_t)BB * NPATCH * CC;
    for (int s = 0; s < 2; ++s) {
      int Hh = 32 << s, Ww = 32 << s, OH = 2 * Hh, OW = 2 * Ww;
      int Mrows = BB * Hh * Ww;
      k_fill_rowmap<<<blocks(4ll * Mrows), 256, 0, stream>>>(RMAP, Hh, Ww, 4 * Mrows);
      for (int dx = 0; dx < 4; ++dx)  // 2x2 transposed conv = 4 rowmapped GEMMs, bias+gelu fused
        gemm(Fs, uptw_bf + ((size_t)(s * 4 + dx)) * CC * CC, uptb + (size_t)s * CC,
             CIN, nullptr, RMAP + dx * Mrows,
             Mrows, CC, CC, CC, CC, CC, 0, 0, 0, 1, 1);
      k_dwconv<<<blocks((long long)BB * OH * OW * CC), 256, 0, stream>>>(
          CIN, updw + (size_t)s * CC * 9, COUT, OH, OW, BB * OH * OW * CC);
      k_layernorm<<<BB * OH * OW, 256, 0, stream>>>(COUT, uplnw + (size_t)s * CC,
                                                    uplnb + (size_t)s * CC, Fn, nullptr);
      Fs = Fn; Fn += (size_t)BB * OH * OW * CC;
    }
    // mask logits: h (100xC) x F2 (16384xC)^T per batch -> d_out
    gemm(HM3, Fs, nullptr,
         outF + (size_t)stage * BB * NQ * 16384, nullptr, nullptr,
         NQ, 16384, CC, CC, CC, 16384,
         (long long)NQ * CC, (long long)16384 * CC, (long long)NQ * 16384, 0, BB);
  };

  // ----- transformer layers -----
  float* Xc = X;
  int Ncur = N1;
  for (int i = 0; i < LL; ++i) {
    if (i == 8) {
      k_insert_q<<<blocks((long long)BB * N2 * CC), 256, 0, stream>>>(Xc, qemb, X2, BB * N2 * CC);
      Xc = X2; Ncur = N2;
    }
    int useMask = 0;
    if (i >= 8) {
      predict(i - 8);
      k_qmask<<<blocks((long long)BB * NQ * 1024), 256, 0, stream>>>(
          outF + (size_t)(i - 8) * BB * NQ * 16384, QM, BB * NQ * 1024);
      useMask = 1;
    }
    int Np = (Ncur + 31) & ~31;          // K-padded attention length (mult of 32)
    int Mr = BB * Ncur;
    k_layernorm<<<Mr, 256, 0, stream>>>(Xc, ln1w + (size_t)i * CC, ln1b + (size_t)i * CC, XN, nullptr);
    gemm(XN, qkvw_bf + (size_t)i * 3 * CC * CC, qkvb + (size_t)i * 3 * CC, TMP0, nullptr, nullptr,
         Mr, 3 * CC, CC, CC, CC, 3 * CC, 0, 0, 0, 0, 1);
    k_qkv_split<<<blocks((long long)Mr * CC), 256, 0, stream>>>(TMP0, Qb, Kbf, Vt, Ncur, Np, Mr * CC);
    k_padvt<<<blocks((long long)BB * NHEADS * DHD * (Np - Ncur)), 256, 0, stream>>>(
        Vt, Ncur, Np, BB * NHEADS * DHD * (Np - Ncur));
    // S = (Q*scale) K^T per (b,h)
    gemm(Qb, Kbf, nullptr, Sbuf, nullptr, nullptr,
         Ncur, Ncur, DHD, DHD, DHD, Np,
         (long long)Ncur * DHD, (long long)Ncur * DHD, (long long)Np * Np, 0, BB * NHEADS);
    k_softmax<<<BB * NHEADS * Ncur, 256, 0, stream>>>(Sbuf, Ab, Ncur, Np, useMask, QM);
    // O = P V  (V stored transposed [d, n]; K = Np, zero-padded)
    gemm(Ab, Vt, nullptr, Obuf, nullptr, nullptr,
         Ncur, DHD, Np, Np, Np, DHD,
         (long long)Np * Np, (long long)DHD * Np, (long long)Ncur * DHD, 0, BB * NHEADS);
    k_merge<<<blocks((long long)Mr * CC), 256, 0, stream>>>(Obuf, ATT, Ncur, Mr * CC);
    gemm(ATT, projw_bf + (size_t)i * CC * CC, projb + (size_t)i * CC, TMP0, nullptr, nullptr,
         Mr, CC, CC, CC, CC, CC, 0, 0, 0, 0, 1);
    k_residual<<<blocks((long long)Mr * CC), 256, 0, stream>>>(Xc, TMP0, ls1 + (size_t)i * CC, Mr * CC);
    k_layernorm<<<Mr, 256, 0, stream>>>(Xc, ln2w + (size_t)i * CC, ln2b + (size_t)i * CC, XN, nullptr);
    gemm(XN, fc1w_bf + (size_t)i * H3 * CC, fc1b + (size_t)i * H3, nullptr, FC1H, nullptr,
         Mr, H3, CC, CC, CC, H3, 0, 0, 0, 1, 1);
    gemm(FC1H, fc2w_bf + (size_t)i * CC * H3, fc2b + (size_t)i * CC, TMP0, nullptr, nullptr,
         Mr, CC, H3, H3, H3, CC, 0, 0, 0, 0, 1);
    k_residual<<<blocks((long long)Mr * CC), 256, 0, stream>>>(Xc, TMP0, ls2 + (size_t)i * CC, Mr * CC);
  }
  predict(4);
}
